// BasicBlock_15436112462307
// MI455X (gfx1250) — 8-bit matrix operand order repaired and checked by reading, compiled, not run on hardware
//
#include <hip/hip_runtime.h>
#include <stdint.h>


typedef __attribute__((ext_vector_type(8))) int   v8i;
typedef __attribute__((ext_vector_type(4))) int   vi4;
typedef __attribute__((ext_vector_type(2))) int   vi2;

#define NB   64
#define NC   256
#define NH   28
#define NW   28
#define NHW  (NH * NW)
#define NP   (NB * NHW)
#define NLEV 7.0f

#define WIN_ROWS   122
#define LDS_STRIDE 272

static const size_t OFF_SC  = 0;
static const size_t OFF_XQ  = 256;
static const size_t XQ_SZ   = (size_t)NP * NC;
static const size_t OFF_WQ1 = OFF_XQ + XQ_SZ;
static const size_t WQ_SZ   = (size_t)9 * NC * NC;
static const size_t OFF_WQ2 = OFF_WQ1 + WQ_SZ;
static const size_t OFF_HQ  = OFF_WQ2 + WQ_SZ;
static const size_t OFF_H   = OFF_HQ + XQ_SZ;

__global__ void init_scalars_kernel(float* sc) {
    if (threadIdx.x < 4) sc[threadIdx.x] = 0.0f;
}

__global__ void absmax_kernel(const float* __restrict__ x, size_t n,
                              float* __restrict__ out) {
    float m = 0.0f;
    for (size_t i = (size_t)blockIdx.x * blockDim.x + threadIdx.x; i < n;
         i += (size_t)gridDim.x * blockDim.x)
        m = fmaxf(m, fabsf(x[i]));
    #pragma unroll
    for (int off = 16; off > 0; off >>= 1)
        m = fmaxf(m, __shfl_xor(m, off, 32));
    __shared__ float sm[8];
    const int lane = threadIdx.x & 31, wv = threadIdx.x >> 5;
    if (lane == 0) sm[wv] = m;
    __syncthreads();
    if (threadIdx.x < 8) {
        float mm = sm[threadIdx.x];
        #pragma unroll
        for (int off = 4; off > 0; off >>= 1)
            mm = fmaxf(mm, __shfl_xor(mm, off, 32));
        if (threadIdx.x == 0)
            atomicMax((unsigned int*)out, __float_as_uint(mm));
    }
}

__global__ void quant_x_kernel(const float* __restrict__ x,
                               const float* __restrict__ amax,
                               int8_t* __restrict__ xq) {
    const size_t tid = (size_t)blockIdx.x * blockDim.x + threadIdx.x;
    if (tid >= (size_t)NP * (NC / 4)) return;
    const int c4 = (int)(tid & 63);
    const int p  = (int)(tid >> 6);
    const int pb = p / NHW, hw = p % NHW;
    const float a = amax[0] + 1e-12f;
    const float s = NLEV / a;
    int packed = 0;
    #pragma unroll
    for (int k = 0; k < 4; ++k) {
        float v = x[((size_t)pb * NC + (c4 * 4 + k)) * NHW + hw];
        v = fminf(a, fmaxf(-a, v));
        const int q = (int)rintf(v * s);
        packed |= (q & 0xff) << (8 * k);
    }
    ((int*)xq)[tid] = packed;
}

__global__ void quant_w_kernel(const float* __restrict__ w,
                               const float* __restrict__ amax,
                               int8_t* __restrict__ wqt) {
    const size_t tid = (size_t)blockIdx.x * blockDim.x + threadIdx.x;
    if (tid >= (size_t)9 * NC * (NC / 4)) return;
    const int i4  = (int)(tid & 63);
    const int o   = (int)((tid >> 6) & 255);
    const int tap = (int)(tid >> 14);
    const float a = amax[0] + 1e-12f;
    const float s = NLEV / a;
    int packed = 0;
    #pragma unroll
    for (int k = 0; k < 4; ++k) {
        float v = w[((size_t)o * NC + (i4 * 4 + k)) * 9 + tap];
        v = fminf(a, fmaxf(-a, v));
        const int q = (int)rintf(v * s);
        packed |= (q & 0xff) << (8 * k);
    }
    ((int*)wqt)[tid] = packed;
}

__global__ void quant_h_kernel(const float* __restrict__ h,
                               const float* __restrict__ amax,
                               int8_t* __restrict__ hq) {
    const size_t tid = (size_t)blockIdx.x * blockDim.x + threadIdx.x;
    if (tid >= (size_t)NP * (NC / 4)) return;
    const float a = amax[0] + 1e-12f;
    const float s = NLEV / a;
    int packed = 0;
    #pragma unroll
    for (int k = 0; k < 4; ++k) {
        float v = h[tid * 4 + k];
        v = fminf(a, fmaxf(-a, v));
        const int q = (int)rintf(v * s);
        packed |= (q & 0xff) << (8 * k);
    }
    ((int*)hq)[tid] = packed;
}

__device__ __forceinline__ v8i load_a_lds(const int8_t* base, bool valid) {
    vi2 c0 = {}, c1 = {}, c2 = {}, c3 = {};
    if (valid) {
        c0 = *(const vi2*)(base +  0);
        c1 = *(const vi2*)(base +  8);
        c2 = *(const vi2*)(base + 32);
        c3 = *(const vi2*)(base + 40);
    }
    v8i a;
    a.s0 = c0.s0; a.s1 = c0.s1; a.s2 = c1.s0; a.s3 = c1.s1;
    a.s4 = c2.s0; a.s5 = c2.s1; a.s6 = c3.s0; a.s7 = c3.s1;
    return a;
}

template <int MODE>
__global__ __launch_bounds__(256) void conv_wmma_kernel(
    const int8_t* __restrict__ aq,
    const int8_t* __restrict__ wqt,
    const float*  __restrict__ sA,
    const float*  __restrict__ sW,
    const float*  __restrict__ gamma, const float* __restrict__ beta,
    const float*  __restrict__ mean,  const float* __restrict__ var,
    const float*  __restrict__ residual,
    float*        __restrict__ outF,
    float*        __restrict__ amax_out)
{
    __shared__ int8_t smem[WIN_ROWS * LDS_STRIDE];

    const int lane  = threadIdx.x & 31;
    const int wave  = threadIdx.x >> 5;
    const int pblk  = blockIdx.x * 64;
    const int p0    = pblk + (wave >> 2) * 32;
    const int n0    = (wave & 3) * 64;
    const int mrow  = lane & 15;
    const int half  = lane >> 4;
    const int nlane = lane & 15;

    const int pwin0 = pblk - 29;
    for (int idx = threadIdx.x; idx < WIN_ROWS * 16; idx += 256) {
        const int row = idx >> 4;
        const int ch  = (idx & 15) * 16;
        const int p   = pwin0 + row;
        vi4 v = {};
        if ((unsigned)p < (unsigned)NP)
            v = *(const vi4*)(aq + (size_t)p * NC + ch);
        *(vi4*)(smem + row * LDS_STRIDE + ch) = v;
    }
    __syncthreads();

    int oh[2], ow[2], wrow[2];
    #pragma unroll
    for (int s = 0; s < 2; ++s) {
        const int p  = p0 + s * 16 + mrow;
        const int hw = p % NHW;
        oh[s]   = hw / NW;
        ow[s]   = hw % NW;
        wrow[s] = p - pwin0;
    }
    const int8_t* bcol = wqt + (size_t)(n0 + nlane) * NC + half * 16;

    v8i acc[2][4] = {};

    for (int tap = 0; tap < 9; ++tap) {
        const int dr = tap / 3 - 1;
        const int ds = tap % 3 - 1;
        const int roff = dr * NW + ds;
        bool val[2];
        const int8_t* abase[2];
        #pragma unroll
        for (int s = 0; s < 2; ++s) {
            val[s] = ((unsigned)(oh[s] + dr) < (unsigned)NH) &&
                     ((unsigned)(ow[s] + ds) < (unsigned)NW);
            abase[s] = smem + (wrow[s] + roff) * LDS_STRIDE + half * 16;
        }
        const int8_t* btap = bcol + (size_t)tap * (NC * NC);

        #pragma unroll
        for (int q = 0; q < 4; ++q) {
            const int ci0 = q * 64;
            const v8i a0 = load_a_lds(abase[0] + ci0, val[0]);
            const v8i a1 = load_a_lds(abase[1] + ci0, val[1]);

            #pragma unroll
            for (int j = 0; j < 4; ++j) {
                const int8_t* bj = btap + ci0 + (size_t)j * 16 * NC;
                vi4 blo = *(const vi4*)(bj);
                vi4 bhi = *(const vi4*)(bj + 32);
                v8i bm;
                bm.s0 = blo.s0; bm.s1 = blo.s1; bm.s2 = blo.s2; bm.s3 = blo.s3;
                bm.s4 = bhi.s0; bm.s5 = bhi.s1; bm.s6 = bhi.s2; bm.s7 = bhi.s3;
                acc[0][j] = __builtin_amdgcn_wmma_i32_16x16x64_iu8(
                    true, a0, true, bm, acc[0][j], false, false);
                acc[1][j] = __builtin_amdgcn_wmma_i32_16x16x64_iu8(
                    true, a1, true, bm, acc[1][j], false, false);
            }
        }
    }

    const float sprod = ((sA[0] + 1e-12f) / NLEV) * ((sW[0] + 1e-12f) / NLEV);
    const int mbase = (lane >> 4) * 8;
    float localmax = 0.0f;

    #pragma unroll
    for (int j = 0; j < 4; ++j) {
        const int c   = n0 + j * 16 + nlane;
        const float inv = gamma[c] * rsqrtf(var[c] + 1e-5f);
        const float bof = beta[c] - mean[c] * inv;
        #pragma unroll
        for (int s = 0; s < 2; ++s) {
            #pragma unroll
            for (int vg = 0; vg < 8; ++vg) {
                const int pp  = p0 + s * 16 + mbase + vg;
                const int bI  = pp / NHW;
                const int hwI = pp % NHW;
                float f = (float)acc[s][j][vg] * sprod;
                if (MODE == 1)
                    f += residual[((size_t)bI * NC + c) * NHW + hwI];
                float t = f * inv + bof;
                t = fminf(1.0f, fmaxf(-1.0f, t));
                if (MODE == 0) {
                    outF[(size_t)pp * NC + c] = t;
                    localmax = fmaxf(localmax, fabsf(t));
                } else {
                    outF[((size_t)bI * NC + c) * NHW + hwI] = t;
                }
            }
        }
    }

    if (MODE == 0) {
        #pragma unroll
        for (int off = 16; off > 0; off >>= 1)
            localmax = fmaxf(localmax, __shfl_xor(localmax, off, 32));
        if (lane == 0)
            atomicMax((unsigned int*)amax_out, __float_as_uint(localmax));
    }
}

extern "C" void kernel_launch(void* const* d_in, const int* in_sizes, int n_in,
                              void* d_out, int out_size, void* d_ws, size_t ws_size,
                              hipStream_t stream) {
    const float* x  = (const float*)d_in[0];
    const float* w1 = (const float*)d_in[1];
    const float* g1 = (const float*)d_in[2];
    const float* b1 = (const float*)d_in[3];
    const float* m1 = (const float*)d_in[4];
    const float* v1 = (const float*)d_in[5];
    const float* w2 = (const float*)d_in[6];
    const float* g2 = (const float*)d_in[7];
    const float* b2 = (const float*)d_in[8];
    const float* m2 = (const float*)d_in[9];
    const float* v2 = (const float*)d_in[10];
    float* out = (float*)d_out;

    char*   ws  = (char*)d_ws;
    float*  sc  = (float*)(ws + OFF_SC);
    int8_t* xq  = (int8_t*)(ws + OFF_XQ);
    int8_t* wq1 = (int8_t*)(ws + OFF_WQ1);
    int8_t* wq2 = (int8_t*)(ws + OFF_WQ2);
    int8_t* hq  = (int8_t*)(ws + OFF_HQ);
    float*  hbuf = (float*)(ws + OFF_H);

    const size_t nX = (size_t)NB * NC * NHW;
    const size_t nW = (size_t)NC * NC * 9;

    init_scalars_kernel<<<1, 32, 0, stream>>>(sc);

    absmax_kernel<<<1024, 256, 0, stream>>>(x,  nX, sc + 0);
    absmax_kernel<<<256,  256, 0, stream>>>(w1, nW, sc + 1);
    absmax_kernel<<<256,  256, 0, stream>>>(w2, nW, sc + 2);

    const int qxBlocks = (int)((nX / 4 + 255) / 256);
    quant_x_kernel<<<qxBlocks, 256, 0, stream>>>(x, sc + 0, xq);
    const int qwBlocks = (int)((nW / 4 + 255) / 256);
    quant_w_kernel<<<qwBlocks, 256, 0, stream>>>(w1, sc + 1, wq1);
    quant_w_kernel<<<qwBlocks, 256, 0, stream>>>(w2, sc + 2, wq2);

    const int mBlocks = NP / 64;
    conv_wmma_kernel<0><<<mBlocks, 256, 0, stream>>>(
        xq, wq1, sc + 0, sc + 1, g1, b1, m1, v1,
        nullptr, hbuf, sc + 3);

    quant_h_kernel<<<qxBlocks, 256, 0, stream>>>(hbuf, sc + 3, hq);

    conv_wmma_kernel<1><<<mBlocks, 256, 0, stream>>>(
        hq, wq2, sc + 3, sc + 2, g2, b2, m2, v2,
        x, out, nullptr);

    (void)in_sizes; (void)n_in; (void)out_size; (void)ws_size;
}
